// MultiHeadAttention_2052994367754
// MI455X (gfx1250) — matrix operand order repaired and checked by reading, not compiled, not run on hardware
//
#include <hip/hip_runtime.h>

// ---------------------------------------------------------------------------
// Shapes (fixed by the reference): B=16, H=8, S=1024.
// Pipeline:
//   1) d  = lrelu(x @ Wd + bd)                      [16,1024]
//   2) h  = lrelu(d @ Wh[z] + bh[z]) per head z     [16,8,1024]
//   3) q/k/v = h(128x1024) @ W* + b*                [128,1024] each
//   4) rank-1 attention: ctx_i = softmax_j(q_i*k_j/32) . v  (per (b,h))
// Workspace: d(64KB) + h,q,k,v(512KB each) ~ 2.1 MB of d_ws.
// Roofline: weights = 48MB (fits 192MB L2); GEMM FLOPs ~1.1GF; dominant
// compute is 134M v_exp_f32 in the rank-1 attention. GEMMs use
// v_wmma_f32_16x16x32_f16 (f32->f16 in regs, f32 accumulate).
// All in-loop offsets are 32-bit so loads use the SADDR+voffset global
// addressing form (scalar base, one 32-bit VGPR offset) instead of 64-bit
// per-lane address pairs.
// ---------------------------------------------------------------------------

typedef __attribute__((ext_vector_type(16))) _Float16 v16h;
typedef __attribute__((ext_vector_type(8)))  float    v8f;

#define LRELU_SLOPE 0.01f

// Generic fused GEMM: C[z] = act( A (MxK, f32) @ W[z] (KxN row-major, f32) + bias[z] )
// One wave32 per block computes one 16x32 C tile (two 16x16 subtiles):
// each K-step issues TWO independent v_wmma_f32_16x16x32_f16 sharing one A
// fragment; K-loop unrolled 2x -> 4 WMMAs in flight per iteration.
__global__ __launch_bounds__(32)
void wmma_gemm_bias_act(const float* __restrict__ A, int lda, long aBatch,
                        const float* __restrict__ W, int N,  long wBatch,
                        const float* __restrict__ bias,      long bBatch,
                        float* __restrict__ C, int ldc,      long cBatch,
                        int K, int act)
{
    const long z = blockIdx.z;

    const int lane = threadIdx.x;          // wave32 lane
    const int hi   = lane >> 4;            // 0 for lanes 0-15, 1 for 16-31
    const int g8   = hi << 3;              // C/D M offset (0/8)
    const int g16  = hi << 4;              // B K-group offset (0/16)
    const int m    = lane & 15;            // A-fragment row (M)
    const int n    = lane & 15;            // B/C-fragment column (N)

    const int mrow0 = blockIdx.y * 16;
    const int ncol0 = blockIdx.x * 32;     // 16x32 C tile per wave

    // Fold batch + per-lane-invariant offsets into the base pointers once;
    // everything inside the K loop is a 32-bit offset (SADDR+voffset form).
    const float* __restrict__ Ap = A + z * aBatch + (mrow0 + m) * lda;
    const float* __restrict__ Wp = W + z * wBatch + (ncol0 + n) + g16 * N;
    const float* __restrict__ bp = bias + z * bBatch;
    float*       __restrict__ Cp = C + z * cBatch;

    v8f acc0 = {};
    v8f acc1 = {};
#pragma unroll 2
    for (int k0 = 0; k0 < K; k0 += 32) {
        v16h af, bf0, bf1;
#pragma unroll
        for (int t = 0; t < 16; ++t) {
            // A 16x32 layout (matches B): K = 16*hi + t
            const int ka = g16 + t;
            af[t] = (_Float16)Ap[k0 + ka];
            // B 32x16 layout (contiguous, matches A): K = 16*hi + t, lane%16 = N
            const int wofs = (k0 + t) * N;
            bf0[t] = (_Float16)Wp[wofs];
            bf1[t] = (_Float16)Wp[wofs + 16];
        }
        if (k0 + 32 < K) {
            // gfx1250 global_prefetch_b8: warm next K-slab of W (both subtiles)
            const int pofs = (k0 + 32) * N;
            __builtin_prefetch(&Wp[pofs], 0, 1);
            __builtin_prefetch(&Wp[pofs + 16], 0, 1);
        }
        // (neg_a, A, neg_b, B, c_mod, C, reuse_a, reuse_b)
        acc0 = __builtin_amdgcn_wmma_f32_16x16x32_f16(false, af, false, bf0,
                                                      (short)0, acc0, false, false);
        acc1 = __builtin_amdgcn_wmma_f32_16x16x32_f16(false, af, false, bf1,
                                                      (short)0, acc1, false, false);
    }

#pragma unroll
    for (int r = 0; r < 8; ++r) {
        // C/D layout: reg r -> M = r + 8*hi, lane%16 -> N
        const int  mm  = mrow0 + r + g8;
        const int  nn0 = ncol0 + n;
        const int  nn1 = nn0 + 16;
        float v0 = acc0[r] + bp[nn0];
        float v1 = acc1[r] + bp[nn1];
        if (act) {
            v0 = (v0 >= 0.0f) ? v0 : LRELU_SLOPE * v0;
            v1 = (v1 >= 0.0f) ? v1 : LRELU_SLOPE * v1;
        }
        Cp[mm * ldc + nn0] = v0;
        Cp[mm * ldc + nn1] = v1;
    }
}

// Rank-1 attention, one block per (b,h). scores never materialized:
//   a = q_i/32 ; m = a>=0 ? a*max(k) : a*min(k)
//   ctx_i = sum_j exp(a*k_j - m)*v_j / sum_j exp(a*k_j - m) ; out = lrelu(ctx)
__global__ __launch_bounds__(256)
void rank1_softmax_attention(const float* __restrict__ q,
                             const float* __restrict__ k,
                             const float* __restrict__ v,
                             float* __restrict__ out)
{
    __shared__ float ks[1024];
    __shared__ float vs[1024];
    __shared__ float red[256];

    const int bh  = blockIdx.x;            // b*H + h, 0..127
    const int tid = threadIdx.x;
    const float* qp = q + bh * 1024;
    const float* kp = k + bh * 1024;
    const float* vp = v + bh * 1024;

    for (int j = tid; j < 1024; j += 256) { ks[j] = kp[j]; vs[j] = vp[j]; }
    __syncthreads();

    float lmax = -3.4e38f, lmin = 3.4e38f;
    for (int j = tid; j < 1024; j += 256) {
        lmax = fmaxf(lmax, ks[j]);
        lmin = fminf(lmin, ks[j]);
    }
    red[tid] = lmax; __syncthreads();
    for (int s = 128; s > 0; s >>= 1) {
        if (tid < s) red[tid] = fmaxf(red[tid], red[tid + s]);
        __syncthreads();
    }
    const float kmax = red[0]; __syncthreads();
    red[tid] = lmin; __syncthreads();
    for (int s = 128; s > 0; s >>= 1) {
        if (tid < s) red[tid] = fminf(red[tid], red[tid + s]);
        __syncthreads();
    }
    const float kmin = red[0]; __syncthreads();

    const float scale = 0.03125f;          // 1/sqrt(1024)
    for (int i = tid; i < 1024; i += 256) {
        const float a    = qp[i] * scale;
        const float mval = (a >= 0.0f) ? a * kmax : a * kmin;
        float se = 0.0f, sev = 0.0f;
#pragma unroll 4
        for (int j = 0; j < 1024; ++j) {
            const float e = __expf(a * ks[j] - mval);   // v_exp_f32
            se  += e;
            sev += e * vs[j];
        }
        const float c = sev / se;
        out[bh * 1024 + i] = (c >= 0.0f) ? c : LRELU_SLOPE * c;
    }
}

extern "C" void kernel_launch(void* const* d_in, const int* in_sizes, int n_in,
                              void* d_out, int out_size, void* d_ws, size_t ws_size,
                              hipStream_t stream)
{
    const float* x  = (const float*)d_in[0];
    const float* Wd = (const float*)d_in[1];
    const float* bd = (const float*)d_in[2];
    const float* Wh = (const float*)d_in[3];
    const float* bh = (const float*)d_in[4];
    const float* Wq = (const float*)d_in[5];
    const float* bq = (const float*)d_in[6];
    const float* Wk = (const float*)d_in[7];
    const float* bk = (const float*)d_in[8];
    const float* Wv = (const float*)d_in[9];
    const float* bv = (const float*)d_in[10];
    float* out = (float*)d_out;

    const int B = 16, H = 8, S = 1024;

    float* ws   = (float*)d_ws;
    float* dbuf = ws;                       // [B,S]
    float* hbuf = dbuf + (long)B * S;       // [B,H,S]
    float* qbuf = hbuf + (long)B * H * S;   // [B*H,S]
    float* kbuf = qbuf + (long)B * H * S;
    float* vbuf = kbuf + (long)B * H * S;

    const dim3 wave(32);

    // 1) d = lrelu(x @ Wd + bd)        grid: 32 N-tiles (16x32 each) x 1 M-tile
    wmma_gemm_bias_act<<<dim3(S / 32, B / 16, 1), wave, 0, stream>>>(
        x, S, 0, Wd, S, 0, bd, 0, dbuf, S, 0, S, 1);

    // 2) h[b,z,t] = lrelu(d @ Wh[z] + bh[z])   grid.z = heads
    wmma_gemm_bias_act<<<dim3(S / 32, B / 16, H), wave, 0, stream>>>(
        dbuf, S, 0, Wh, S, (long)S * S, bh, S, hbuf, H * S, S, S, 1);

    // 3) q/k/v = h(128xS) @ W + b (no activation)
    wmma_gemm_bias_act<<<dim3(S / 32, (B * H) / 16, 1), wave, 0, stream>>>(
        hbuf, S, 0, Wq, S, 0, bq, 0, qbuf, S, 0, S, 0);
    wmma_gemm_bias_act<<<dim3(S / 32, (B * H) / 16, 1), wave, 0, stream>>>(
        hbuf, S, 0, Wk, S, 0, bk, 0, kbuf, S, 0, S, 0);
    wmma_gemm_bias_act<<<dim3(S / 32, (B * H) / 16, 1), wave, 0, stream>>>(
        hbuf, S, 0, Wv, S, 0, bv, 0, vbuf, S, 0, S, 0);

    // 4) rank-1 softmax attention + final lrelu -> d_out [B,H,S]
    rank1_softmax_attention<<<dim3(B * H), dim3(256), 0, stream>>>(
        qbuf, kbuf, vbuf, out);
}
